// GAT_59193239273527
// MI455X (gfx1250) — matrix operand order repaired and checked by reading, not compiled, not run on hardware
//
#include <hip/hip_runtime.h>
#include <hip/hip_bf16.h>
#include <cstdint>
#include <cstddef>

typedef __attribute__((ext_vector_type(16))) __bf16 v16bf;
typedef __attribute__((ext_vector_type(8)))  __bf16 v8bf;
typedef __attribute__((ext_vector_type(8)))  float  v8f;

namespace {
constexpr int kN    = 20000;   // nodes
constexpr int kInF  = 256;
constexpr int kHid  = 128;
constexpr int kH    = 4;       // heads
constexpr int kOutF = 10;
constexpr int kE    = 400000;  // edges (without self loops)
constexpr int kG    = 64;      // graphs
constexpr int kHC   = kHid * kH;      // 512
constexpr int kE2   = kE + kN;        // edges + self loops = 420000
}

// ---------------------------------------------------------------------------
// Convert 16 fp32 (4 float4) into one 16-half bf16 WMMA fragment.
// ---------------------------------------------------------------------------
__device__ inline v16bf cvt_frag(float4 f0, float4 f1, float4 f2, float4 f3)
{
    v16bf a;
    a[0]  = (__bf16)f0.x; a[1]  = (__bf16)f0.y; a[2]  = (__bf16)f0.z; a[3]  = (__bf16)f0.w;
    a[4]  = (__bf16)f1.x; a[5]  = (__bf16)f1.y; a[6]  = (__bf16)f1.z; a[7]  = (__bf16)f1.w;
    a[8]  = (__bf16)f2.x; a[9]  = (__bf16)f2.y; a[10] = (__bf16)f2.z; a[11] = (__bf16)f2.w;
    a[12] = (__bf16)f3.x; a[13] = (__bf16)f3.y; a[14] = (__bf16)f3.z; a[15] = (__bf16)f3.w;
    return a;
}

// ---------------------------------------------------------------------------
// WMMA GEMM:  C[M,Nn] = act(A[M,K] @ W[K,Nn] + bias)
// 256 threads = 8 waves; each wave owns a 32(M) x 64(N) tile:
//   2 A fragments x 4 B fragments -> 8 v_wmma_f32_16x16x32_bf16 per k-step.
// A loads are software-pipelined one k-iteration ahead; W tile (K x 64) is
// staged once in LDS as bf16, column-major (K contiguous per column).
// Requires: M % 32 == 0, Nn % 64 == 0, K % 32 == 0 (all true here).
// ---------------------------------------------------------------------------
__global__ void __launch_bounds__(256)
gemm_bias_act(const float* __restrict__ A, const float* __restrict__ W,
              const float* __restrict__ bias, float* __restrict__ C,
              int M, int K, int Nn, int act)
{
    extern __shared__ char smem_raw[];
    __bf16* wlds = (__bf16*)smem_raw;           // [64 cols][K] column-major

    const int tid = threadIdx.x;
    const int n0  = blockIdx.y * 64;

    // cooperative, coalesced load of the 64-wide weight tile, fp32 -> bf16
    for (int idx = tid; idx < K * 64; idx += 256) {
        int k = idx >> 6, n = idx & 63;
        wlds[n * K + k] = (__bf16)W[(size_t)k * Nn + n0 + n];
    }
    __syncthreads();

    const int wave  = tid >> 5;
    const int lane  = tid & 31;
    const int mtile = blockIdx.x * 8 + wave;     // 32-row tile index
    if (mtile * 32 >= M) return;                 // wave-uniform exit

    const int mrow0 = mtile * 32 + (lane & 15);
    const int kbA   = (lane >> 4) * 8;           // A frag: K 0-7/16-23 vs 8-15/24-31
    const int kbB   = (lane >> 4) * 8;           // B frag: K 0-7/16-23 vs 8-15/24-31
    const int nB    = lane & 15;

    v8f acc[8];
    for (int s = 0; s < 8; ++s)
        for (int j = 0; j < 8; ++j) acc[s][j] = 0.0f;

    const float* pa0 = A + (size_t)mrow0 * K + kbA;        // row mrow0
    const float* pa1 = pa0 + (size_t)16 * K;               // row mrow0 + 16

    // prologue: first iteration's A data
    float4 f0 = *(const float4*)(pa0);
    float4 f1 = *(const float4*)(pa0 + 4);
    float4 f2 = *(const float4*)(pa0 + 16);
    float4 f3 = *(const float4*)(pa0 + 20);
    float4 f4 = *(const float4*)(pa1);
    float4 f5 = *(const float4*)(pa1 + 4);
    float4 f6 = *(const float4*)(pa1 + 16);
    float4 f7 = *(const float4*)(pa1 + 20);

    for (int k0 = 0; k0 < K; k0 += 32) {
        // convert current A data to fragments (frees f0..f7 for the next loads)
        v16bf a0 = cvt_frag(f0, f1, f2, f3);
        v16bf a1 = cvt_frag(f4, f5, f6, f7);

        // issue next iteration's A loads now; they retire behind the WMMAs
        if (k0 + 32 < K) {
            const float* q0 = pa0 + k0 + 32;
            const float* q1 = pa1 + k0 + 32;
            __builtin_prefetch(q0 + 64, 0, 3);   // +2 iterations, toward WGP
            __builtin_prefetch(q1 + 64, 0, 3);
            f0 = *(const float4*)(q0);
            f1 = *(const float4*)(q0 + 4);
            f2 = *(const float4*)(q0 + 16);
            f3 = *(const float4*)(q0 + 20);
            f4 = *(const float4*)(q1);
            f5 = *(const float4*)(q1 + 4);
            f6 = *(const float4*)(q1 + 16);
            f7 = *(const float4*)(q1 + 20);
        }

        // all 4 B fragments before the WMMA chain (staged dscnt waits)
        v16bf bf[4];
        #pragma unroll
        for (int s = 0; s < 4; ++s) {
            const __bf16* col = &wlds[(size_t)(s * 16 + nB) * K + k0 + kbB];
            v8bf lo = *(const v8bf*)col;
            v8bf hi = *(const v8bf*)(col + 16);
            #pragma unroll
            for (int j = 0; j < 8; ++j) { bf[s][j] = lo[j]; bf[s][8 + j] = hi[j]; }
        }

        #pragma unroll
        for (int s = 0; s < 4; ++s)
            acc[s] = __builtin_amdgcn_wmma_f32_16x16x32_bf16(
                false, a0, false, bf[s], (short)0, acc[s], false, false);
        #pragma unroll
        for (int s = 0; s < 4; ++s)
            acc[4 + s] = __builtin_amdgcn_wmma_f32_16x16x32_bf16(
                false, a1, false, bf[s], (short)0, acc[4 + s], false, false);
    }

    const int rbase = 8 * (lane >> 4);
    #pragma unroll
    for (int half = 0; half < 2; ++half) {
        #pragma unroll
        for (int s = 0; s < 4; ++s) {
            int col  = n0 + s * 16 + nB;
            float bv = bias[col];
            #pragma unroll
            for (int r = 0; r < 8; ++r) {
                int row = mtile * 32 + half * 16 + rbase + r;
                float v = acc[half * 4 + s][r] + bv;
                if (act) v = v > 0.0f ? v : (__expf(v) - 1.0f);   // ELU
                C[(size_t)row * Nn + col] = v;
            }
        }
    }
}

// ---------------------------------------------------------------------------
// Edge / softmax / scatter kernels (memory-bound phase, fp32)
// ---------------------------------------------------------------------------
__device__ inline void atomicMaxF(float* addr, float val)
{
    unsigned int* ua  = (unsigned int*)addr;
    unsigned int  cur = __float_as_uint(*addr);
    while (__uint_as_float(cur) < val) {
        unsigned int prev = atomicCAS(ua, cur, __float_as_uint(val));
        if (prev == cur) break;
        cur = prev;
    }
}

__global__ void init_md(float* mmax, float* denom, int nh)
{
    int idx = blockIdx.x * blockDim.x + threadIdx.x;
    if (idx < nh)            mmax[idx] = -1e30f;
    else if (idx < 2 * nh)   denom[idx - nh] = 0.0f;
}

__global__ void init_out(float* out, const float* __restrict__ bias, size_t n)
{
    size_t idx = (size_t)blockIdx.x * blockDim.x + threadIdx.x;
    if (idx < n) out[idx] = bias[idx & (kHC - 1)];
}

// one wave32 per edge: logits[e][h] = att_h . LeakyReLU(xl[src] + xr[dst])
__global__ void edge_scores(const float* __restrict__ xl, const float* __restrict__ xr,
                            const float* __restrict__ att, const int* __restrict__ ei,
                            float* __restrict__ logits, float* mmax)
{
    int wid  = (blockIdx.x * blockDim.x + threadIdx.x) >> 5;
    int lane = threadIdx.x & 31;
    if (wid >= kE2) return;                       // wave-uniform
    int s = (wid < kE) ? ei[wid]      : (wid - kE);
    int d = (wid < kE) ? ei[kE + wid] : (wid - kE);
    const float* xls = xl + (size_t)s * kHC;
    const float* xrd = xr + (size_t)d * kHC;
    #pragma unroll
    for (int h = 0; h < kH; ++h) {
        float acc = 0.0f;
        #pragma unroll
        for (int i = 0; i < kHid / 32; ++i) {
            int c   = i * 32 + lane;
            float v = xls[h * kHid + c] + xrd[h * kHid + c];
            v = v > 0.0f ? v : 0.2f * v;          // LeakyReLU(0.2)
            acc += v * att[h * kHid + c];
        }
        for (int off = 16; off > 0; off >>= 1) acc += __shfl_xor(acc, off, 32);
        if (lane == 0) {
            logits[(size_t)wid * kH + h] = acc;
            atomicMaxF(&mmax[d * kH + h], acc);
        }
    }
}

// p = exp(logit - max[dst]); denom[dst] += p   (overwrites logits with p)
__global__ void edge_softmax(const int* __restrict__ ei, const float* __restrict__ mmax,
                             float* logits, float* denom)
{
    int idx = blockIdx.x * blockDim.x + threadIdx.x;
    if (idx >= kE2 * kH) return;
    int e = idx >> 2, h = idx & 3;
    int d = (e < kE) ? ei[kE + e] : (e - kE);
    float p = __expf(logits[idx] - mmax[d * kH + h]);
    logits[idx] = p;
    atomicAdd(&denom[d * kH + h], p);
}

// one wave32 per edge: out[dst] += alpha * xl[src]
__global__ void edge_aggregate(const float* __restrict__ xl, const float* __restrict__ p,
                               const float* __restrict__ denom, const int* __restrict__ ei,
                               float* out)
{
    int wid  = (blockIdx.x * blockDim.x + threadIdx.x) >> 5;
    int lane = threadIdx.x & 31;
    if (wid >= kE2) return;
    int s = (wid < kE) ? ei[wid]      : (wid - kE);
    int d = (wid < kE) ? ei[kE + wid] : (wid - kE);
    const float* xls = xl + (size_t)s * kHC;
    float*       od  = out + (size_t)d * kHC;
    #pragma unroll
    for (int h = 0; h < kH; ++h) {
        float alpha = p[(size_t)wid * kH + h] / denom[d * kH + h];
        #pragma unroll
        for (int i = 0; i < kHid / 32; ++i) {
            int c = h * kHid + i * 32 + lane;
            atomicAdd(&od[c], alpha * xls[c]);
        }
    }
}

__global__ void elu_k(const float* __restrict__ in, float* __restrict__ out, size_t n)
{
    size_t idx = (size_t)blockIdx.x * blockDim.x + threadIdx.x;
    if (idx >= n) return;
    float v = in[idx];
    out[idx] = v > 0.0f ? v : (__expf(v) - 1.0f);
}

// ---------------------------------------------------------------------------
// Pooling + classifier
// ---------------------------------------------------------------------------
__global__ void zero_k(float* p, int n)
{
    int idx = blockIdx.x * blockDim.x + threadIdx.x;
    if (idx < n) p[idx] = 0.0f;
}

__global__ void pool_scatter(const float* __restrict__ h, const int* __restrict__ batch,
                             float* sums, float* cnts)
{
    size_t idx = (size_t)blockIdx.x * blockDim.x + threadIdx.x;
    if (idx >= (size_t)kN * kHC) return;
    int n = (int)(idx >> 9);          // / 512
    int c = (int)(idx & (kHC - 1));
    int g = batch[n];
    atomicAdd(&sums[(size_t)g * kHC + c], h[idx]);
    if (c == 0) atomicAdd(&cnts[g], 1.0f);
}

__global__ void classify(const float* __restrict__ sums, const float* __restrict__ cnts,
                         const float* __restrict__ W, const float* __restrict__ b,
                         float* out)
{
    int idx = blockIdx.x * blockDim.x + threadIdx.x;
    if (idx >= kG * kOutF) return;
    int g = idx / kOutF, o = idx % kOutF;
    float inv = 1.0f / fmaxf(cnts[g], 1.0f);
    float acc = b[o];
    for (int c = 0; c < kHC; ++c)
        acc += sums[(size_t)g * kHC + c] * inv * W[c * kOutF + o];
    out[idx] = acc;
}

// ---------------------------------------------------------------------------
extern "C" void kernel_launch(void* const* d_in, const int* in_sizes, int n_in,
                              void* d_out, int out_size, void* d_ws, size_t ws_size,
                              hipStream_t stream)
{
    const float* x     = (const float*)d_in[0];
    const int*   ei    = (const int*)d_in[1];
    const int*   batch = (const int*)d_in[2];
    const float* preW  = (const float*)d_in[3];
    const float* preb  = (const float*)d_in[4];
    const float* clsW  = (const float*)d_in[5];
    const float* clsb  = (const float*)d_in[6];

    float* ws = (float*)d_ws;
    const size_t NHC = (size_t)kN * kHC;
    float* hA     = ws;                               // activations (input of layer)
    float* hB     = hA + NHC;                         // aggregation accumulator
    float* xlb    = hB + NHC;
    float* xrb    = xlb + NHC;
    float* logits = xrb + NHC;                        // E2*H, reused as p
    float* mmax   = logits + (size_t)kE2 * kH;
    float* denom  = mmax + (size_t)kN * kH;
    float* sums   = denom + (size_t)kN * kH;          // G*HC (cnts right after)
    float* cnts   = sums + (size_t)kG * kHC;

    const dim3 blk(256);
    const int  mblocks = (kN / 32 + 7) / 8;           // 79 blocks of 8 waves

    // pre-projection: hA = elu(x @ preW + preb)
    gemm_bias_act<<<dim3(mblocks, kHid / 64), blk, (size_t)64 * kInF * sizeof(__bf16),
                    stream>>>(x, preW, preb, hA, kN, kInF, kHid, 1);

    int curK = kHid;
    for (int l = 0; l < 3; ++l) {
        const float* Wl   = (const float*)d_in[7 + l * 6 + 0];
        const float* bl   = (const float*)d_in[7 + l * 6 + 1];
        const float* Wr   = (const float*)d_in[7 + l * 6 + 2];
        const float* br   = (const float*)d_in[7 + l * 6 + 3];
        const float* att  = (const float*)d_in[7 + l * 6 + 4];
        const float* biasv= (const float*)d_in[7 + l * 6 + 5];

        size_t sh = (size_t)64 * curK * sizeof(__bf16);
        gemm_bias_act<<<dim3(mblocks, kHC / 64), blk, sh, stream>>>(
            hA, Wl, bl, xlb, kN, curK, kHC, 0);
        gemm_bias_act<<<dim3(mblocks, kHC / 64), blk, sh, stream>>>(
            hA, Wr, br, xrb, kN, curK, kHC, 0);

        init_md<<<(2 * kN * kH + 255) / 256, blk, 0, stream>>>(mmax, denom, kN * kH);
        init_out<<<(int)((NHC + 255) / 256), blk, 0, stream>>>(hB, biasv, NHC);

        edge_scores<<<(kE2 + 7) / 8, blk, 0, stream>>>(xlb, xrb, att, ei, logits, mmax);
        edge_softmax<<<(kE2 * kH + 255) / 256, blk, 0, stream>>>(ei, mmax, logits, denom);
        edge_aggregate<<<(kE2 + 7) / 8, blk, 0, stream>>>(xlb, logits, denom, ei, hB);

        elu_k<<<(int)((NHC + 255) / 256), blk, 0, stream>>>(hB, hA, NHC);
        curK = kHC;
    }

    zero_k<<<(kG * kHC + kG + 255) / 256, blk, 0, stream>>>(sums, kG * kHC + kG);
    pool_scatter<<<(int)((NHC + 255) / 256), blk, 0, stream>>>(hA, batch, sums, cnts);
    classify<<<(kG * kOutF + 255) / 256, blk, 0, stream>>>(sums, cnts, clsW, clsb,
                                                           (float*)d_out);
}
